// MatrixAttention_21869973471317
// MI455X (gfx1250) — matrix operand order repaired and checked by reading, not compiled, not run on hardware
//
#include <hip/hip_runtime.h>
#include <hip/hip_bf16.h>

typedef __attribute__((ext_vector_type(16))) _Float16 v16h;
typedef __attribute__((ext_vector_type(8)))  float    v8f;
typedef __attribute__((ext_vector_type(4)))  unsigned int v4u;
typedef __attribute__((ext_vector_type(8)))  int      v8i;
typedef __attribute__((ext_vector_type(4)))  int      v4i;

#define H5     65
#define HW2    (H5*H5)            // 4225
#define NHD    8                  // heads
#define KDIM   8
#define HDD    16
#define DIM    64
#define HIDDEN 128
#define CRCV   (HIDDEN + 4*NHD*KDIM)   // 384
#define ASCALE 0.35355339059327373f    // 8^-0.5

// ---- WMMA 16x16x32 f16 fragment index maps used by this code (wave32, contiguous K) ----
// A: lane l holds row M=l&15; element e -> K = e + 16*(l>>4)
// B: lane l holds col N=l&15; element e -> K = e + 16*(l>>4)
// C: VGPR r holds M = r + 8*(l>>4), N = l&15
__device__ __forceinline__ int a_kof(int e, int hi) {
  return e + 16*hi;
}

// ===================== weight packing (A-fragment order) =====================
__global__ void pack_w_rcv(const float* __restrict__ w, _Float16* __restrict__ ap) {
  int blk = blockIdx.x;                 // mt*36 + ks
  int ks  = blk % 36;
  int lane = threadIdx.x, hi = lane >> 4, lo = lane & 15;
  int M = (blk / 36) * 16 + lo;
  _Float16* dst = ap + ((size_t)blk * 32 + lane) * 16;
  #pragma unroll
  for (int e = 0; e < 16; ++e) {
    int K = ks * 32 + a_kof(e, hi);     // K = (kh*3+kw)*128 + ic
    int khw = K >> 7, ic = K & 127;
    int kh = khw / 3, kw = khw % 3;
    dst[e] = (_Float16)w[(((size_t)M * HIDDEN + ic) * 3 + kh) * 3 + kw];
  }
}

__global__ void pack_w_out(const float* __restrict__ w, _Float16* __restrict__ ap) {
  int blk = blockIdx.x;                 // mt*18 + ks
  int ks  = blk % 18;
  int lane = threadIdx.x, hi = lane >> 4, lo = lane & 15;
  int M = (blk / 18) * 16 + lo;
  _Float16* dst = ap + ((size_t)blk * 32 + lane) * 16;
  #pragma unroll
  for (int e = 0; e < 16; ++e) {
    int K = ks * 32 + a_kof(e, hi);     // K = (kh*3+kw)*64 + ic
    int khw = K >> 6, ic = K & 63;
    int kh = khw / 3, kw = khw % 3;
    dst[e] = (_Float16)w[(((size_t)M * DIM + ic) * 3 + kh) * 3 + kw];
  }
}

// pack v[b,H,d,i,j] (from z) into A-fragments per (b,H,i): M=d, K=j (padded to 96)
__global__ void pack_v(const float* __restrict__ z, _Float16* __restrict__ vp) {
  int ks = blockIdx.x, i = blockIdx.y, bh = blockIdx.z;
  int b = bh >> 3, H = bh & 7;
  int lane = threadIdx.x, hi = lane >> 4, d = lane & 15;
  _Float16* dst = vp + ((((size_t)bh * H5 + i) * 3 + ks) * 32 + lane) * 16;
  const float* src = z + (((size_t)b * CRCV + H * 48 + 32 + d) * H5 + i) * H5;
  #pragma unroll
  for (int e = 0; e < 16; ++e) {
    int j = ks * 32 + a_kof(e, hi);
    dst[e] = (_Float16)((j < H5) ? src[j] : 0.f);
  }
}

// ===================== in_proj: grouped k2 s2 p1 conv + BN -> NHWC f16 ======
__global__ void in_proj(const float* __restrict__ x, const float* __restrict__ w,
                        const float* __restrict__ s, const float* __restrict__ bb,
                        _Float16* __restrict__ yb) {
  int idx = blockIdx.x * 256 + threadIdx.x;
  if (idx >= 4 * HIDDEN * HW2) return;
  int ow = idx % H5, oh = (idx / H5) % H5;
  int oc = (idx / HW2) % HIDDEN, b = idx / (HW2 * HIDDEN);
  const float* xp = x + ((size_t)b * DIM + (oc >> 1)) * 128 * 128;
  float acc = 0.f;
  for (int kh = 0; kh < 2; ++kh) {
    int ih = 2 * oh - 1 + kh; if (ih < 0 || ih >= 128) continue;
    for (int kw = 0; kw < 2; ++kw) {
      int iw = 2 * ow - 1 + kw; if (iw < 0 || iw >= 128) continue;
      acc += w[(oc * 2 + kh) * 2 + kw] * xp[ih * 128 + iw];
    }
  }
  acc = acc * s[oc] + bb[oc];
  yb[(((size_t)b * H5 + oh) * H5 + ow) * HIDDEN + oc] = (_Float16)acc;
}

// ============ rcv: implicit-GEMM 3x3 conv 128->384, M=384 N=16 K=1152 =======
// B-tile staged via the Tensor Data Mover for interior blocks (one 2D D#:
// 3 rows x 2304 f16, row stride 65*128 elems), manual staging on borders
// where zero-padding is required.
__global__ __launch_bounds__(256) void rcv_gemm(
    const _Float16* __restrict__ yb, const _Float16* __restrict__ ap,
    const float* __restrict__ s, const float* __restrict__ bb,
    float* __restrict__ z) {
  __shared__ _Float16 sB[3][18][HIDDEN];           // [ih][iw][ic], K-contiguous
  int owt = blockIdx.x, oh = blockIdx.y, b = blockIdx.z;
  int tid = threadIdx.x;

  bool interior = (oh >= 1) && (oh <= 63) && (owt >= 1) && (owt <= 3);
  if (interior) {
    if (tid < 32) {                                // one wave issues the TDM op
      size_t gbase = (size_t)(const void*)(yb +
          (((size_t)b * H5 + (oh - 1)) * H5 + (owt * 16 - 1)) * HIDDEN);
      unsigned lds = (unsigned)(uintptr_t)(void*)&sB[0][0][0];
      unsigned dim0 = H5 * HIDDEN;                 // tensor row length (elems)
      unsigned dim1 = H5;                          // tensor rows
      unsigned til0 = 18 * HIDDEN;                 // tile row length (elems)
      unsigned til1 = 3;                           // tile rows
      unsigned long long st0 = (unsigned long long)H5 * HIDDEN;
      v4u g0 = {};
      g0[0] = 1u;                                  // count=1, user mode
      g0[1] = lds;                                 // lds_addr
      g0[2] = (unsigned)gbase;                     // global_addr[31:0]
      g0[3] = (unsigned)((gbase >> 32) & 0x01FFFFFFu) | 0x80000000u; // [56:32]|type=2
      v8i g1 = {};
      g1[0] = 0x10000;                             // data_size=1 (2 bytes)
      g1[1] = (int)((dim0 & 0xFFFFu) << 16);       // tensor_dim0 lo
      g1[2] = (int)((dim0 >> 16) | ((dim1 & 0xFFFFu) << 16));
      g1[3] = (int)((dim1 >> 16) | (til0 << 16));  // tile_dim0
      g1[4] = (int)til1;                           // tile_dim1 (tile_dim2=0)
      g1[5] = (int)(st0 & 0xFFFFFFFFu);            // tensor_dim0_stride lo
      g1[6] = (int)((st0 >> 32) & 0xFFFFu);        // stride hi, dim1_stride=0
      g1[7] = 0;
      v4i g2 = {}, g3 = {};
      v8i g4 = {};                                 // unused trailing group (zeros)
      __builtin_amdgcn_tensor_load_to_lds(g0, g1, g2, g3, g4, 0);
      __builtin_amdgcn_s_wait_tensorcnt(0);
    }
  } else {
    for (int t = tid; t < 3 * 18 * HIDDEN; t += 256) {
      int ic = t & 127, iw2 = (t >> 7) % 18, ih2 = t / (18 * 128);
      int gh = oh - 1 + ih2, gw = owt * 16 - 1 + iw2;
      _Float16 v = (_Float16)0.f;
      if (gh >= 0 && gh < H5 && gw >= 0 && gw < H5)
        v = yb[(((size_t)b * H5 + gh) * H5 + gw) * HIDDEN + ic];
      (&sB[0][0][0])[t] = v;
    }
  }
  __syncthreads();

  int wave = tid >> 5, lane = tid & 31;
  int hi = lane >> 4, lo = lane & 15;
  v8f acc0 = {}, acc1 = {}, acc2 = {};
  const _Float16* ap0 = ap + (size_t)wave * 36 * 512 + lane * 16;
  const _Float16* ap1 = ap0 + (size_t)8 * 36 * 512;
  const _Float16* ap2 = ap0 + (size_t)16 * 36 * 512;
  for (int ks = 0; ks < 36; ++ks) {
    size_t o = (size_t)ks * 512;
    __builtin_prefetch(ap0 + o + 512, 0, 3);       // WGP-scope stream prefetch
    int khw = ks >> 2, ic0 = (ks & 3) * 32 + 16 * hi;
    int kh = khw / 3, kw = khw % 3;
    v16h bf = *(const v16h*)(&sB[kh][kw + lo][ic0]);
    v16h a0 = *(const v16h*)(ap0 + o);
    v16h a1 = *(const v16h*)(ap1 + o);
    v16h a2 = *(const v16h*)(ap2 + o);
    acc0 = __builtin_amdgcn_wmma_f32_16x16x32_f16(false, a0, false, bf,
                                                  (short)0, acc0, false, false);
    acc1 = __builtin_amdgcn_wmma_f32_16x16x32_f16(false, a1, false, bf,
                                                  (short)0, acc1, false, false);
    acc2 = __builtin_amdgcn_wmma_f32_16x16x32_f16(false, a2, false, bf,
                                                  (short)0, acc2, false, false);
  }
  int ow = owt * 16 + lo;
  if (ow < H5) {
    float* zp = z + (size_t)b * CRCV * HW2 + (size_t)oh * H5 + ow;
    #pragma unroll
    for (int r = 0; r < 8; ++r) {
      int oc = wave * 16 + r + 8 * hi;
      zp[(size_t)oc * HW2] = acc0[r] * s[oc] + bb[oc];
    }
    #pragma unroll
    for (int r = 0; r < 8; ++r) {
      int oc = (wave + 8) * 16 + r + 8 * hi;
      zp[(size_t)oc * HW2] = acc1[r] * s[oc] + bb[oc];
    }
    #pragma unroll
    for (int r = 0; r < 8; ++r) {
      int oc = (wave + 16) * 16 + r + 8 * hi;
      zp[(size_t)oc * HW2] = acc2[r] * s[oc] + bb[oc];
    }
  }
}

// ===================== row / column attention softmax =======================
__global__ __launch_bounds__(256) void row_softmax(const float* __restrict__ z,
                                                   float* __restrict__ r) {
  __shared__ float sq[KDIM][H5], sk[KDIM][H5], sl[H5][H5];
  int w = blockIdx.x, H = blockIdx.y, b = blockIdx.z;
  int tid = threadIdx.x;
  const float* zb = z + (size_t)b * CRCV * HW2;
  for (int t = tid; t < KDIM * H5; t += 256) {
    int d = t / H5, i = t % H5;
    sq[d][i] = zb[((size_t)(H * 48 + d) * H5 + i) * H5 + w];
    sk[d][i] = zb[((size_t)(H * 48 + 8 + d) * H5 + i) * H5 + w];
  }
  __syncthreads();
  for (int p = tid; p < HW2; p += 256) {
    int i = p / H5, j = p % H5;
    float a = 0.f;
    #pragma unroll
    for (int d = 0; d < KDIM; ++d) a += sq[d][i] * sk[d][j];
    sl[i][j] = a * ASCALE;
  }
  __syncthreads();
  if (tid < H5) {
    int j = tid;
    float m = -1e30f;
    for (int i = 0; i < H5; ++i) m = fmaxf(m, sl[i][j]);
    float sum = 0.f;
    for (int i = 0; i < H5; ++i) { float e = __expf(sl[i][j] - m); sl[i][j] = e; sum += e; }
    float inv = 1.f / sum;
    for (int i = 0; i < H5; ++i)
      r[(((size_t)(b * NHD + H) * H5 + i) * H5 + j) * H5 + w] = sl[i][j] * inv;
  }
}

__global__ __launch_bounds__(256) void col_softmax(const float* __restrict__ z,
                                                   float* __restrict__ c) {
  __shared__ float sq[KDIM][H5], sk[KDIM][H5], sl[H5][H5];
  int h = blockIdx.x, H = blockIdx.y, b = blockIdx.z;
  int tid = threadIdx.x;
  const float* zb = z + (size_t)b * CRCV * HW2;
  for (int t = tid; t < KDIM * H5; t += 256) {
    int d = t / H5, i = t % H5;
    sq[d][i] = zb[((size_t)(H * 48 + 16 + d) * H5 + h) * H5 + i];
    sk[d][i] = zb[((size_t)(H * 48 + 24 + d) * H5 + h) * H5 + i];
  }
  __syncthreads();
  for (int p = tid; p < HW2; p += 256) {
    int i = p / H5, j = p % H5;   // i = source col, j = output col
    float a = 0.f;
    #pragma unroll
    for (int d = 0; d < KDIM; ++d) a += sq[d][i] * sk[d][j];
    sl[i][j] = a * ASCALE;
  }
  __syncthreads();
  if (tid < H5) {
    int j = tid;
    float m = -1e30f;
    for (int i = 0; i < H5; ++i) m = fmaxf(m, sl[i][j]);
    float sum = 0.f;
    for (int i = 0; i < H5; ++i) { float e = __expf(sl[i][j] - m); sl[i][j] = e; sum += e; }
    float inv = 1.f / sum;
    for (int i = 0; i < H5; ++i)
      c[(((size_t)(b * NHD + H) * H5 + i) * H5 + h) * H5 + j] = sl[i][j] * inv;
  }
}

// ====== combine: attn(d,h,w) = sum_i r(i,h,w) * [ v(d,i,:) @ c(:,h,w) ] =====
__global__ __launch_bounds__(160) void combine(
    const _Float16* __restrict__ vp, const float* __restrict__ cbuf,
    const float* __restrict__ rbuf, float* __restrict__ attn) {
  __shared__ _Float16 sC[80][96];          // [w][j] K-contiguous, zero padded
  __shared__ float    sR[H5][80];
  int h = blockIdx.x, H = blockIdx.y, b = blockIdx.z;
  int bh = b * NHD + H;
  int tid = threadIdx.x;
  for (int t = tid; t < 80 * 96; t += 160) {
    int w = t / 96, j = t % 96;
    float v = 0.f;
    if (w < H5 && j < H5)
      v = cbuf[(((size_t)bh * H5 + j) * H5 + h) * H5 + w];
    sC[w][j] = (_Float16)v;
  }
  for (int t = tid; t < H5 * 80; t += 160) {
    int i = t / 80, w = t % 80;
    sR[i][w] = (w < H5) ? rbuf[(((size_t)bh * H5 + i) * H5 + h) * H5 + w] : 0.f;
  }
  __syncthreads();
  int nt = tid >> 5, lane = tid & 31;      // 5 waves = 5 N-tiles of w
  int hi = lane >> 4, lo = lane & 15;
  int w = nt * 16 + lo;
  float acc[8] = {};
  const _Float16* vpb = vp + (size_t)bh * H5 * 1536 + lane * 16;
  for (int i = 0; i < H5; ++i) {
    v8f u = {};
    const _Float16* vpi = vpb + (size_t)i * 1536;
    __builtin_prefetch(vpi + 1536, 0, 3);
    #pragma unroll
    for (int ks = 0; ks < 3; ++ks) {
      v16h a  = *(const v16h*)(vpi + (size_t)ks * 512);
      v16h bf = *(const v16h*)(&sC[w][ks * 32 + 16 * hi]);
      u = __builtin_amdgcn_wmma_f32_16x16x32_f16(false, a, false, bf,
                                                 (short)0, u, false, false);
    }
    float rw = sR[i][w];
    #pragma unroll
    for (int q = 0; q < 8; ++q) acc[q] += rw * u[q];
  }
  if (w < H5) {
    #pragma unroll
    for (int q = 0; q < 8; ++q) {
      int d = q + 8 * hi;
      attn[(((size_t)b * HIDDEN + H * HDD + d) * H5 + h) * H5 + w] = acc[q];
    }
  }
}

// ============ positional-encoding branch: grouped 3x3 (2 in-ch/group) on v, += attn =======
__global__ void pe_add(const float* __restrict__ z, const float* __restrict__ w,
                       const float* __restrict__ s, const float* __restrict__ bb,
                       float* __restrict__ attn) {
  int idx = blockIdx.x * 256 + threadIdx.x;
  if (idx >= 4 * HIDDEN * HW2) return;
  int ww = idx % H5, h = (idx / H5) % H5;
  int oc = (idx / HW2) % HIDDEN, b = idx / (HW2 * HIDDEN);
  int g = oc >> 1;
  float acc = 0.f;
  for (int icg = 0; icg < 2; ++icg) {
    int c = 2 * g + icg;
    int zc = (c >> 4) * 48 + 32 + (c & 15);      // vr channel -> z channel
    const float* vpp = z + ((size_t)b * CRCV + zc) * HW2;
    for (int kh = 0; kh < 3; ++kh) {
      int ih = h - 1 + kh; if (ih < 0 || ih >= H5) continue;
      for (int kw = 0; kw < 3; ++kw) {
        int iw = ww - 1 + kw; if (iw < 0 || iw >= H5) continue;
        acc += w[((oc * 2 + icg) * 3 + kh) * 3 + kw] * vpp[ih * H5 + iw];
      }
    }
  }
  attn[idx] += acc * s[oc] + bb[oc];
}

// ====== grouped conv-transpose k2 s2 p1 (parity-resolved) -> NHWC f16 =======
__global__ void dconv(const float* __restrict__ attn, const float* __restrict__ w,
                      const float* __restrict__ bias, _Float16* __restrict__ dcb) {
  int idx = blockIdx.x * 256 + threadIdx.x;
  if (idx >= 4 * DIM * 128 * 128) return;
  int ow = idx % 128, oh = (idx / 128) % 128;
  int oc = (idx / 16384) % DIM, b = idx / (16384 * DIM);
  int kh = oh & 1, kw = ow & 1;
  int ihd = (oh + kh) >> 1, iwd = (ow + kw) >> 1;
  float acc = bias[oc];
  #pragma unroll
  for (int icg = 0; icg < 2; ++icg)
    acc += w[((oc * 2 + icg) * 2 + kh) * 2 + kw] *
           attn[(((size_t)b * HIDDEN + 2 * oc + icg) * H5 + ihd) * H5 + iwd];
  dcb[(((size_t)b * 128 + oh) * 128 + ow) * DIM + oc] = (_Float16)acc;
}

// ============ out_proj[1]: implicit-GEMM 3x3 conv 64->64 + BN ===============
__global__ __launch_bounds__(128) void out_gemm(
    const _Float16* __restrict__ dcb, const _Float16* __restrict__ ap,
    const float* __restrict__ s, const float* __restrict__ bb,
    float* __restrict__ out) {
  __shared__ _Float16 sB[3][18][DIM];
  int owt = blockIdx.x, oh = blockIdx.y, b = blockIdx.z;
  int tid = threadIdx.x;
  for (int t = tid; t < 3 * 18 * DIM; t += 128) {
    int ic = t & 63, iw2 = (t >> 6) % 18, ih2 = t / (18 * 64);
    int gh = oh - 1 + ih2, gw = owt * 16 - 1 + iw2;
    _Float16 v = (_Float16)0.f;
    if (gh >= 0 && gh < 128 && gw >= 0 && gw < 128)
      v = dcb[(((size_t)b * 128 + gh) * 128 + gw) * DIM + ic];
    (&sB[0][0][0])[t] = v;
  }
  __syncthreads();
  int mt = tid >> 5, lane = tid & 31;      // 4 waves = 4 M-tiles
  int hi = lane >> 4, lo = lane & 15;
  v8f acc = {};
  const _Float16* apm = ap + (size_t)mt * 18 * 512 + lane * 16;
  for (int ks = 0; ks < 18; ++ks) {
    size_t o = (size_t)ks * 512;
    __builtin_prefetch(apm + o + 512, 0, 3);
    int khw = ks >> 1, ic0 = (ks & 1) * 32 + 16 * hi;
    int kh = khw / 3, kw = khw % 3;
    v16h a = *(const v16h*)(apm + o);
    v16h bf = *(const v16h*)(&sB[kh][kw + lo][ic0]);
    acc = __builtin_amdgcn_wmma_f32_16x16x32_f16(false, a, false, bf,
                                                 (short)0, acc, false, false);
  }
  int ow = owt * 16 + lo;
  #pragma unroll
  for (int r = 0; r < 8; ++r) {
    int oc = mt * 16 + r + 8 * hi;
    out[(((size_t)b * DIM + oc) * 128 + oh) * 128 + ow] = acc[r] * s[oc] + bb[oc];
  }
}

// ============================ launch ========================================
static constexpr size_t al256(size_t x) { return (x + 255) & ~(size_t)255; }

extern "C" void kernel_launch(void* const* d_in, const int* in_sizes, int n_in,
                              void* d_out, int out_size, void* d_ws, size_t ws_size,
                              hipStream_t stream) {
  const float* x     = (const float*)d_in[0];
  const float* w_in  = (const float*)d_in[1];
  const float* s_in  = (const float*)d_in[2];
  const float* b_in  = (const float*)d_in[3];
  const float* w_rcv = (const float*)d_in[4];
  const float* s_rcv = (const float*)d_in[5];
  const float* b_rcv = (const float*)d_in[6];
  const float* w_pe  = (const float*)d_in[7];
  const float* s_pe  = (const float*)d_in[8];
  const float* b_pe  = (const float*)d_in[9];
  const float* w_dc  = (const float*)d_in[10];
  const float* b_dc  = (const float*)d_in[11];
  const float* w_out = (const float*)d_in[12];
  const float* s_out = (const float*)d_in[13];
  const float* b_out = (const float*)d_in[14];

  char* ws = (char*)d_ws;
  constexpr size_t OFF_YB   = 0;
  constexpr size_t OFF_Z    = al256(OFF_YB   + (size_t)4 * HW2 * HIDDEN * 2);
  constexpr size_t OFF_R    = al256(OFF_Z    + (size_t)4 * CRCV * HW2 * 4);
  constexpr size_t OFF_C    = al256(OFF_R    + (size_t)32 * H5 * HW2 * 4);
  constexpr size_t OFF_ATT  = al256(OFF_C    + (size_t)32 * H5 * HW2 * 4);
  constexpr size_t OFF_DCB  = al256(OFF_ATT  + (size_t)4 * HIDDEN * HW2 * 4);
  constexpr size_t OFF_ARCV = al256(OFF_DCB  + (size_t)4 * 128 * 128 * DIM * 2);
  constexpr size_t OFF_AOUT = al256(OFF_ARCV + (size_t)24 * 36 * 512 * 2);
  constexpr size_t OFF_VP   = al256(OFF_AOUT + (size_t)4 * 18 * 512 * 2);

  _Float16* yb   = (_Float16*)(ws + OFF_YB);
  float*    z    = (float*)   (ws + OFF_Z);
  float*    rbuf = (float*)   (ws + OFF_R);
  float*    cbuf = (float*)   (ws + OFF_C);
  float*    attn = (float*)   (ws + OFF_ATT);
  _Float16* dcb  = (_Float16*)(ws + OFF_DCB);
  _Float16* arcv = (_Float16*)(ws + OFF_ARCV);
  _Float16* aout = (_Float16*)(ws + OFF_AOUT);
  _Float16* vp   = (_Float16*)(ws + OFF_VP);

  pack_w_rcv <<<24 * 36, 32, 0, stream>>>(w_rcv, arcv);
  pack_w_out <<<4 * 18,  32, 0, stream>>>(w_out, aout);
  in_proj    <<<(4 * HIDDEN * HW2 + 255) / 256, 256, 0, stream>>>(x, w_in, s_in, b_in, yb);
  rcv_gemm   <<<dim3(5, H5, 4), 256, 0, stream>>>(yb, arcv, s_rcv, b_rcv, z);
  pack_v     <<<dim3(3, H5, 32), 32, 0, stream>>>(z, vp);
  row_softmax<<<dim3(H5, NHD, 4), 256, 0, stream>>>(z, rbuf);
  col_softmax<<<dim3(H5, NHD, 4), 256, 0, stream>>>(z, cbuf);
  combine    <<<dim3(H5, NHD, 4), 160, 0, stream>>>(vp, cbuf, rbuf, attn);
  pe_add     <<<(4 * HIDDEN * HW2 + 255) / 256, 256, 0, stream>>>(z, w_pe, s_pe, b_pe, attn);
  dconv      <<<(4 * DIM * 128 * 128 + 255) / 256, 256, 0, stream>>>(attn, w_dc, b_dc, dcb);
  out_gemm   <<<dim3(8, 128, 4), 128, 0, stream>>>(dcb, aout, s_out, b_out, (float*)d_out);
}
